// Block_43250320671483
// MI455X (gfx1250) — matrix operand order repaired and checked by reading, not compiled, not run on hardware
//
#include <hip/hip_runtime.h>
#include <cstdint>
#include <cstddef>

// ---------------------------------------------------------------------------
// Types for CDNA5 WMMA (wave32, 16x16x32 bf16 -> f32)
// ---------------------------------------------------------------------------
typedef __bf16 bf16;
typedef __attribute__((ext_vector_type(16))) bf16  v16bf;
typedef __attribute__((ext_vector_type(8)))  float v8f;

#define Bn   4
#define Tn   2048
#define Cn   768
#define Hn   12
#define HDn  64
#define BTn  (Bn * Tn)
#define C4n  (4 * Cn)
#define NQKV (3 * Cn)   // 2304

// ---------------------------------------------------------------------------
// WMMA helpers
// ---------------------------------------------------------------------------
static __device__ __forceinline__ v8f wmma_bf16(v16bf a, v16bf b, v8f c) {
  // D = A(16x32) * B(32x16) + C, fp32 accumulate
  return __builtin_amdgcn_wmma_f32_16x16x32_bf16(
      /*neg_a=*/false, a, /*neg_b=*/false, b,
      /*c_mod=*/(short)0, c, /*reuse_a=*/false, /*reuse_b=*/false);
}

// A-matrix 16x32 bf16 fragment. Lane layout used here (contiguous k, same order as B):
//   lanes 0-15 : row m = lane,    k = 0..15
//   lanes16-31 : row m = lane-16, k = 16..31
// -> two contiguous 16B runs at [m][klo] and [m][klo+8], klo = 16*(lane>=16)
static __device__ __forceinline__ v16bf load_frag_a(const bf16* base, int stride,
                                                    int m, int klo) {
  const uint4* p0 = reinterpret_cast<const uint4*>(base + (size_t)m * stride + klo);
  const uint4* p1 = reinterpret_cast<const uint4*>(base + (size_t)m * stride + klo + 8);
  union { v16bf v; uint4 u[2]; } t;
  t.u[0] = *p0; t.u[1] = *p1;
  return t.v;
}

// B-matrix 32x16 bf16 fragment, source stored "transposed" row-major [N][K]:
//   lanes 0-15 : col n = lane,    k = 0..15
//   lanes16-31 : col n = lane-16, k = 16..31
// -> one contiguous 32B run at [n][k0], k0 = 16*(lane>=16)
static __device__ __forceinline__ v16bf load_frag_b(const bf16* base, int stride,
                                                    int n, int k0) {
  const uint4* p = reinterpret_cast<const uint4*>(base + (size_t)n * stride + k0);
  union { v16bf v; uint4 u[2]; } t;
  t.u[0] = p[0]; t.u[1] = p[1];
  return t.v;
}

// B fragment from a [K][N] tile (used for V in attention): strided gather.
static __device__ __forceinline__ v16bf load_frag_b_strided(const bf16* base,
                                                            int rowstride,
                                                            int k0, int n) {
  v16bf r;
#pragma unroll
  for (int e = 0; e < 16; ++e) r[e] = base[(size_t)(k0 + e) * rowstride + n];
  return r;
}

// ---------------------------------------------------------------------------
// CDNA5 async global->LDS copy (GLOBAL_LOAD_ASYNC_TO_LDS_B128, ASYNCcnt).
// The low 32 bits of a generic pointer to __shared__ are the LDS byte offset
// (see ISA 10.2, LDS aperture).
// ---------------------------------------------------------------------------
static __device__ __forceinline__ void async_copy_b128(const bf16* gsrc,
                                                       bf16* lds_dst) {
  uint32_t la = (uint32_t)(uintptr_t)lds_dst;
  uint64_t ga = (uint64_t)(uintptr_t)gsrc;
  asm volatile("global_load_async_to_lds_b128 %0, %1, off"
               :: "v"(la), "v"(ga) : "memory");
}
static __device__ __forceinline__ void wait_async0() {
  asm volatile("s_wait_asynccnt 0" ::: "memory");
}

// ---------------------------------------------------------------------------
// Weight conversion kernels (fp32 -> bf16, transposed to [N][K])
// ---------------------------------------------------------------------------
__global__ void transpose_convert_kernel(const float* __restrict__ in,
                                         bf16* __restrict__ out, int K, int N) {
  int idx = blockIdx.x * blockDim.x + threadIdx.x;
  if (idx >= K * N) return;
  int n = idx / K, k = idx - n * K;
  out[(size_t)n * K + k] = (bf16)in[(size_t)k * N + n];
}

// wq/wk/wv [H][C][HD] -> wqkvT [3*C][C] with n = h*64+d (q cols 0..767, etc.)
__global__ void convert_qkv_w_kernel(const float* __restrict__ wq,
                                     const float* __restrict__ wk,
                                     const float* __restrict__ wv,
                                     bf16* __restrict__ out) {
  int idx = blockIdx.x * blockDim.x + threadIdx.x;
  if (idx >= 3 * Cn * Cn) return;
  int s = idx / (Cn * Cn);
  int r = idx - s * (Cn * Cn);
  int n = r / Cn, k = r - n * Cn;
  int h = n >> 6, d = n & 63;
  const float* w = (s == 0) ? wq : (s == 1) ? wk : wv;
  out[idx] = (bf16)w[((size_t)h * Cn + k) * HDn + d];
}

__global__ void pack_qkv_bias_kernel(const float* __restrict__ bq,
                                     const float* __restrict__ bk,
                                     const float* __restrict__ bv,
                                     float* __restrict__ out) {
  int i = blockIdx.x * blockDim.x + threadIdx.x;
  if (i >= NQKV) return;
  out[i] = (i < Cn) ? bq[i] : (i < 2 * Cn) ? bk[i - Cn] : bv[i - 2 * Cn];
}

// ---------------------------------------------------------------------------
// LayerNorm: fp32 in -> bf16 out, one row (C=768) per 256-thread block
// ---------------------------------------------------------------------------
__global__ __launch_bounds__(256) void ln_kernel(const float* __restrict__ x,
                                                 const float* __restrict__ g,
                                                 const float* __restrict__ bta,
                                                 bf16* __restrict__ out) {
  __shared__ float red[256];
  const int row = blockIdx.x, tid = threadIdx.x;
  const float* xr = x + (size_t)row * Cn;
  float v0 = xr[tid], v1 = xr[tid + 256], v2 = xr[tid + 512];
  float s = v0 + v1 + v2;
  red[tid] = s; __syncthreads();
#pragma unroll
  for (int o = 128; o > 0; o >>= 1) { if (tid < o) red[tid] += red[tid + o]; __syncthreads(); }
  const float mu = red[0] * (1.0f / Cn);
  __syncthreads();
  float d0 = v0 - mu, d1 = v1 - mu, d2 = v2 - mu;
  red[tid] = d0 * d0 + d1 * d1 + d2 * d2; __syncthreads();
#pragma unroll
  for (int o = 128; o > 0; o >>= 1) { if (tid < o) red[tid] += red[tid + o]; __syncthreads(); }
  const float rstd = rsqrtf(red[0] * (1.0f / Cn) + 1e-5f);
  bf16* orow = out + (size_t)row * Cn;
  orow[tid]       = (bf16)(d0 * rstd * g[tid]       + bta[tid]);
  orow[tid + 256] = (bf16)(d1 * rstd * g[tid + 256] + bta[tid + 256]);
  orow[tid + 512] = (bf16)(d2 * rstd * g[tid + 512] + bta[tid + 512]);
}

// ---------------------------------------------------------------------------
// bf16 WMMA GEMM: out[M,N] = A[M,K] @ Bt[N,K]^T (+bias) (+resid) (relu?)
// WG tile 64x256, 8 waves (2x4) each computing 32x64 via 2x4 accumulators.
// Double-buffered LDS filled by async global->LDS copies: while the matrix
// pipe runs 16 WMMAs on stage k, stage k+1 streams in (ASYNCcnt overlap);
// one s_wait_asynccnt + one barrier per 64-deep K stage.
// ---------------------------------------------------------------------------
template <typename OutT, bool RELU>
__global__ __launch_bounds__(256) void gemm_bf16_kernel(
    const bf16* __restrict__ A, const bf16* __restrict__ Bt,
    const float* __restrict__ bias, const float* __restrict__ resid,
    OutT* __restrict__ out, int M, int N, int K) {
  __shared__ __align__(16) bf16 As[2][64][72];     // 2 x 64 x 64 (+8 pad)
  __shared__ __align__(16) bf16 Bs[2][256][72];    // 2 x 256 x 64 (+8 pad)
  const int tid = threadIdx.x;
  const int wid = tid >> 5, lane = tid & 31;
  const int lm = lane & 15, lhalf = lane & 16, mof = lhalf >> 1;  // mof: 0|8
  const int mbase = blockIdx.y * 64, nbase = blockIdx.x * 256;
  const int wm = wid >> 2, wn = wid & 3;           // wave tile: 32 x 64
  v8f acc[2][4] = {};

  auto issue_stage = [&](int buf, int kb) {
    // A: 64x64 -> 512 x b128, 2 per thread ; B: 256x64 -> 2048 x b128, 8/thread
#pragma unroll
    for (int i = 0; i < 2; ++i) {
      int idx = tid + i * 256;
      int r = idx >> 3, c8 = (idx & 7) * 8;
      async_copy_b128(&A[(size_t)(mbase + r) * K + kb + c8], &As[buf][r][c8]);
    }
#pragma unroll
    for (int i = 0; i < 8; ++i) {
      int idx = tid + i * 256;
      int r = idx >> 3, c8 = (idx & 7) * 8;
      async_copy_b128(&Bt[(size_t)(nbase + r) * K + kb + c8], &Bs[buf][r][c8]);
    }
  };

  issue_stage(0, 0);
  int cur = 0;
  for (int kb = 0; kb < K; kb += 64) {
    wait_async0();        // own copies done...
    __syncthreads();      // ...then everyone's: stage `cur` is fully in LDS
    if (kb + 64 < K) issue_stage(cur ^ 1, kb + 64);   // overlap next fill

#pragma unroll
    for (int kc = 0; kc < 2; ++kc) {
      v16bf af[2];
#pragma unroll
      for (int mi = 0; mi < 2; ++mi)
        af[mi] = load_frag_a(&As[cur][0][0], 72, wm * 32 + mi * 16 + lm, kc * 32 + lhalf);
#pragma unroll
      for (int ni = 0; ni < 4; ++ni) {
        v16bf bfr = load_frag_b(&Bs[cur][0][0], 72, wn * 64 + ni * 16 + lm, kc * 32 + lhalf);
#pragma unroll
        for (int mi = 0; mi < 2; ++mi)
          acc[mi][ni] = wmma_bf16(af[mi], bfr, acc[mi][ni]);
      }
    }
    cur ^= 1;
  }

#pragma unroll
  for (int mi = 0; mi < 2; ++mi)
#pragma unroll
    for (int ni = 0; ni < 4; ++ni)
#pragma unroll
      for (int r = 0; r < 8; ++r) {
        int m = mbase + wm * 32 + mi * 16 + r + mof;
        int n = nbase + wn * 64 + ni * 16 + lm;
        float v = acc[mi][ni][r];
        if (bias)  v += bias[n];
        if (resid) v += resid[(size_t)m * N + n];
        if (RELU)  v = fmaxf(v, 0.0f);
        out[(size_t)m * N + n] = (OutT)v;
      }
}

// ---------------------------------------------------------------------------
// Flash attention, causal, NO 1/sqrt(hd) scaling (matches reference).
// qkv: bf16 [B*T][2304] (q|k|v, col = h*64+d).  attnb: bf16 [B*T][768].
// grid = (B*H, T/128); 8 waves; wave w handles 16 query rows.
// K/V tiles double-buffered via async global->LDS copies.
// ---------------------------------------------------------------------------
__global__ __launch_bounds__(256) void attn_kernel(const bf16* __restrict__ qkv,
                                                   bf16* __restrict__ attnb) {
  __shared__ __align__(16) bf16 Ks[2][32][72];    // 32 keys x 64 d (padded)
  __shared__ __align__(16) bf16 Vs[2][32][72];
  __shared__ __align__(16) bf16 Pb[8][16][48];    // per-wave P transpose bounce
  const int tid = threadIdx.x, wid = tid >> 5, lane = tid & 31;
  const int lm = lane & 15, lhalf = lane & 16, mof = lhalf >> 1;
  const int bh = blockIdx.x;
  const int b = bh / Hn, h = bh - b * Hn;
  const int qb0 = blockIdx.y * 128;
  const int qbase = qb0 + wid * 16;
  const size_t ld = (size_t)NQKV;

  // Q fragments (16 rows x 64 d = two k-chunks), loaded straight from global.
  const bf16* qrow = qkv + ((size_t)(b * Tn) + (qbase + lm)) * ld + h * HDn;
  v16bf qf[2];
#pragma unroll
  for (int kc = 0; kc < 2; ++kc) qf[kc] = load_frag_a(qrow, 0, 0, kc * 32 + lhalf);

  float mrow[8], lrow[8];
  v8f O[4] = {};
#pragma unroll
  for (int r = 0; r < 8; ++r) { mrow[r] = -INFINITY; lrow[r] = 0.0f; }

  auto issue_kv = [&](int buf, int jb) {
    int r = tid >> 3, c8 = (tid & 7) * 8;
    const size_t src = ((size_t)(b * Tn) + (jb + r)) * ld + h * HDn + c8;
    async_copy_b128(qkv + src + Cn,     &Ks[buf][r][c8]);
    async_copy_b128(qkv + src + 2 * Cn, &Vs[buf][r][c8]);
  };

  const int jend = qb0 + 128;           // causal bound for this WG
  issue_kv(0, 0);
  int cur = 0;
  for (int jb = 0; jb < jend; jb += 32) {
    wait_async0();
    __syncthreads();
    if (jb + 32 < jend) issue_kv(cur ^ 1, jb + 32);

    // --- scores S = Q @ K^T for 32 keys (two 16-col tiles) ---
    float s[2][8];
#pragma unroll
    for (int nt = 0; nt < 2; ++nt) {
      v8f accS = {};
      accS = wmma_bf16(qf[0], load_frag_b(&Ks[cur][0][0], 72, nt * 16 + lm, 0  + lhalf), accS);
      accS = wmma_bf16(qf[1], load_frag_b(&Ks[cur][0][0], 72, nt * 16 + lm, 32 + lhalf), accS);
      const int keyg = jb + nt * 16 + lm;
#pragma unroll
      for (int r = 0; r < 8; ++r) {
        int qg = qbase + r + mof;
        s[nt][r] = (keyg <= qg) ? accS[r] : -INFINITY;
      }
    }

    // --- online softmax (row reductions across 16-lane half-groups) ---
    float alpha[8];
#pragma unroll
    for (int r = 0; r < 8; ++r) {
      float tmax = fmaxf(s[0][r], s[1][r]);
#pragma unroll
      for (int off = 1; off < 16; off <<= 1)
        tmax = fmaxf(tmax, __shfl_xor(tmax, off, 16));
      float mn = fmaxf(mrow[r], tmax);
      alpha[r] = __expf(mrow[r] - mn);
      mrow[r] = mn;
      float p0 = __expf(s[0][r] - mn), p1 = __expf(s[1][r] - mn);
      s[0][r] = p0; s[1][r] = p1;
      float rs = p0 + p1;
#pragma unroll
      for (int off = 1; off < 16; off <<= 1) rs += __shfl_xor(rs, off, 16);
      lrow[r] = lrow[r] * alpha[r] + rs;
    }
#pragma unroll
    for (int ct = 0; ct < 4; ++ct)
#pragma unroll
      for (int r = 0; r < 8; ++r) O[ct][r] *= alpha[r];

    // --- P: C-layout -> A-layout via per-wave LDS bounce ---
#pragma unroll
    for (int nt = 0; nt < 2; ++nt)
#pragma unroll
      for (int r = 0; r < 8; ++r)
        Pb[wid][r + mof][nt * 16 + lm] = (bf16)s[nt][r];
    asm volatile("s_wait_dscnt 0" ::: "memory");
    v16bf pf = load_frag_a(&Pb[wid][0][0], 48, lm, lhalf);

    // --- O += P(16x32) @ V(32x64) ---
#pragma unroll
    for (int ct = 0; ct < 4; ++ct) {
      v16bf vf = load_frag_b_strided(&Vs[cur][0][0], 72, lhalf, ct * 16 + lm);
      O[ct] = wmma_bf16(pf, vf, O[ct]);
    }
    cur ^= 1;
  }

  // --- finalize: O / l, store concat-head layout [B*T][768] ---
#pragma unroll
  for (int r = 0; r < 8; ++r) {
    const float inv = 1.0f / lrow[r];
    const int tq = qbase + r + mof;
    bf16* orow = attnb + ((size_t)(b * Tn) + tq) * Cn + h * HDn;
#pragma unroll
    for (int ct = 0; ct < 4; ++ct)
      orow[ct * 16 + lm] = (bf16)(O[ct][r] * inv);
  }
}

// ---------------------------------------------------------------------------
// Host-side orchestration
// ---------------------------------------------------------------------------
extern "C" void kernel_launch(void* const* d_in, const int* in_sizes, int n_in,
                              void* d_out, int out_size, void* d_ws, size_t ws_size,
                              hipStream_t stream) {
  const float* x      = (const float*)d_in[0];
  const float* wq     = (const float*)d_in[1];
  const float* bq     = (const float*)d_in[2];
  const float* wk     = (const float*)d_in[3];
  const float* bk     = (const float*)d_in[4];
  const float* wv     = (const float*)d_in[5];
  const float* bv     = (const float*)d_in[6];
  const float* w_proj = (const float*)d_in[7];
  const float* b_proj = (const float*)d_in[8];
  const float* w1     = (const float*)d_in[9];
  const float* b1     = (const float*)d_in[10];
  const float* w2     = (const float*)d_in[11];
  const float* b2     = (const float*)d_in[12];
  const float* g1     = (const float*)d_in[13];
  const float* beta1  = (const float*)d_in[14];
  const float* g2     = (const float*)d_in[15];
  const float* beta2  = (const float*)d_in[16];
  float* out = (float*)d_out;

  uint8_t* wsb = (uint8_t*)d_ws;
  size_t off = 0;
  auto alloc = [&](size_t bytes) -> void* {
    void* p = wsb + off;
    off = (off + bytes + 255) & ~(size_t)255;
    return p;
  };
  bf16*  wqkvT  = (bf16*)alloc((size_t)3 * Cn * Cn * sizeof(bf16));
  bf16*  wprojT = (bf16*)alloc((size_t)Cn * Cn * sizeof(bf16));
  bf16*  w1T    = (bf16*)alloc((size_t)C4n * Cn * sizeof(bf16));
  bf16*  w2T    = (bf16*)alloc((size_t)Cn * C4n * sizeof(bf16));
  float* biasQ  = (float*)alloc((size_t)NQKV * sizeof(float));
  bf16*  hb     = (bf16*)alloc((size_t)BTn * Cn * sizeof(bf16));
  bf16*  qkvb   = (bf16*)alloc((size_t)BTn * NQKV * sizeof(bf16));
  bf16*  attnb  = (bf16*)alloc((size_t)BTn * Cn * sizeof(bf16));
  float* x2     = (float*)alloc((size_t)BTn * Cn * sizeof(float));
  bf16*  h2b    = (bf16*)alloc((size_t)BTn * Cn * sizeof(bf16));
  bf16*  ub     = (bf16*)alloc((size_t)BTn * C4n * sizeof(bf16));
  (void)ws_size; (void)in_sizes; (void)n_in; (void)out_size;

  auto blocks = [](long n) { return (unsigned)((n + 255) / 256); };

  // 1) weight/bias conversion (bf16, transposed to [N][K])
  convert_qkv_w_kernel<<<blocks(3L * Cn * Cn), 256, 0, stream>>>(wq, wk, wv, wqkvT);
  pack_qkv_bias_kernel<<<blocks(NQKV), 256, 0, stream>>>(bq, bk, bv, biasQ);
  transpose_convert_kernel<<<blocks((long)Cn * Cn), 256, 0, stream>>>(w_proj, wprojT, Cn, Cn);
  transpose_convert_kernel<<<blocks((long)Cn * C4n), 256, 0, stream>>>(w1, w1T, Cn, C4n);
  transpose_convert_kernel<<<blocks((long)C4n * Cn), 256, 0, stream>>>(w2, w2T, C4n, Cn);

  // 2) LN1 -> hb (bf16)
  ln_kernel<<<BTn, 256, 0, stream>>>(x, g1, beta1, hb);

  // 3) fused QKV projection: [8192,768] x [768,2304] -> qkvb (bf16, +bias)
  gemm_bf16_kernel<bf16, false><<<dim3(NQKV / 256, BTn / 64), 256, 0, stream>>>(
      hb, wqkvT, biasQ, nullptr, qkvb, BTn, NQKV, Cn);

  // 4) causal flash attention -> attnb (bf16, heads concatenated)
  attn_kernel<<<dim3(Bn * Hn, Tn / 128), 256, 0, stream>>>(qkvb, attnb);

  // 5) output projection + residual: x2 = x + attnb @ w_proj + b_proj (fp32)
  gemm_bf16_kernel<float, false><<<dim3(Cn / 256, BTn / 64), 256, 0, stream>>>(
      attnb, wprojT, b_proj, x, x2, BTn, Cn, Cn);

  // 6) LN2 -> h2b (bf16)
  ln_kernel<<<BTn, 256, 0, stream>>>(x2, g2, beta2, h2b);

  // 7) MLP up + ReLU: ub = relu(h2b @ w1 + b1) (bf16)
  gemm_bf16_kernel<bf16, true><<<dim3(C4n / 256, BTn / 64), 256, 0, stream>>>(
      h2b, w1T, b1, nullptr, ub, BTn, C4n, Cn);

  // 8) MLP down + residual: out = x2 + ub @ w2 + b2 (fp32)
  gemm_bf16_kernel<float, false><<<dim3(Cn / 256, BTn / 64), 256, 0, stream>>>(
      ub, w2T, b2, x2, out, BTn, Cn, C4n);
}
